// MultiHeadAttention_44023414784180
// MI455X (gfx1250) — matrix operand order repaired and checked by reading, not compiled, not run on hardware
//
#include <hip/hip_runtime.h>

typedef __attribute__((ext_vector_type(16))) _Float16 v16h;
typedef __attribute__((ext_vector_type(8)))  _Float16 v8h;
typedef __attribute__((ext_vector_type(8)))  float    v8f;
typedef __attribute__((ext_vector_type(4)))  unsigned int u32x4;
typedef __attribute__((ext_vector_type(8)))  int      i32x8;
typedef __attribute__((ext_vector_type(4)))  int      i32x4;

#define D_MODEL 1024
#define NUM_HEADS 16
#define DK 64
#define BATCH 2
#define SEQ 2048
#define MTOK (BATCH * SEQ) /* 4096 */

#define CH 128             /* K-chunk staged in LDS per TDM transfer (halves) */
#define NCH (D_MODEL / CH) /* 8 chunks */
#define LDSTR 136          /* padded LDS row stride in halves (128 + 8)       */

#if __has_builtin(__builtin_amdgcn_tensor_load_to_lds) && \
    __has_builtin(__builtin_amdgcn_s_wait_tensorcnt)
#define USE_TDM 1
#endif

// ---------------------------------------------------------------------------
// WMMA helpers (gfx1250, wave32). Fragment layouts (C/D per CDNA5 ISA 7.12.2; A/B use this file's contiguous K order):
//  A (16x32 f16):  lane = 16*half + m ; element i <-> K = i + 16*half
//  B (32x16 f16):  lane = 16*half + n ; element i <-> K = i + 16*half
//  C/D (16x16 f32): lane = 16*half + n ; element r <-> M = r + 8*half
// ---------------------------------------------------------------------------
__device__ __forceinline__ v8f wmma_f16(v16h a, v16h b, v8f c) {
  return __builtin_amdgcn_wmma_f32_16x16x32_f16(
      /*neg_a=*/false, a, /*neg_b=*/false, b,
      /*c_mod=*/(short)0, c, /*reuse_a=*/false, /*reuse_b=*/false);
}

// A fragment from a row-major f16 matrix (row stride ld halves).
__device__ __forceinline__ v16h load_a_frag(const _Float16* __restrict__ base,
                                            int row, int ld, int k0) {
  int lane = threadIdx.x & 31;
  int m = lane & 15, hl = lane >> 4;
  const _Float16* p = base + (size_t)(row + m) * ld + k0 + 16 * hl;
  v8h lo = *(const v8h*)(p);
  v8h hi = *(const v8h*)(p + 8);
  v16h a;
#pragma unroll
  for (int i = 0; i < 8; ++i) { a[i] = lo[i]; a[i + 8] = hi[i]; }
  return a;
}

// A fragment from the staged LDS tile (row stride LDSTR halves), K-step s (0..3).
__device__ __forceinline__ v16h load_a_frag_lds(const _Float16* lp, int s) {
  int lane = threadIdx.x & 31;
  int m = lane & 15, hl = lane >> 4;
  const _Float16* p = lp + m * LDSTR + s * 32 + 16 * hl;
  v8h lo = *(const v8h*)(p);
  v8h hi = *(const v8h*)(p + 8);
  v16h a;
#pragma unroll
  for (int i = 0; i < 8; ++i) { a[i] = lo[i]; a[i + 8] = hi[i]; }
  return a;
}

// B fragment from pre-packed weights: Wpack[kt][n][32], contiguous 32B per lane.
__device__ __forceinline__ v16h load_b_frag_packed(const _Float16* __restrict__ wp,
                                                   int kt, int ncol) {
  int lane = threadIdx.x & 31;
  int n = lane & 15, hl = lane >> 4;
  return *(const v16h*)(wp + (size_t)kt * (32 * D_MODEL) + (size_t)(ncol + n) * 32 + 16 * hl);
}

// B fragment for S = Q*K^T: B[d][j] = K[j][d]; K row-major [t][DK] -> contiguous.
__device__ __forceinline__ v16h load_b_frag_kt(const _Float16* __restrict__ kh,
                                               int j0, int k0) {
  int lane = threadIdx.x & 31;
  int j = lane & 15, hl = lane >> 4;
  return *(const v16h*)(kh + (size_t)(j0 + j) * DK + k0 + 16 * hl);
}

// B fragment for O = P*V: B[j][d] = Vt[d][j]; Vt row-major [DK][SEQ] -> contiguous.
__device__ __forceinline__ v16h load_b_frag_vt(const _Float16* __restrict__ vt,
                                               int d0, int j0) {
  int lane = threadIdx.x & 31;
  int d = lane & 15, hl = lane >> 4;
  return *(const v16h*)(vt + (size_t)(d0 + d) * SEQ + j0 + 16 * hl);
}

// ---------------------------------------------------------------------------
// Tensor Data Mover: async-load one 16xCH f16 tile (row stride D_MODEL halves)
// into LDS with 8-half row padding (-> LDSTR).  D# packing per ISA 08 §8.
// ---------------------------------------------------------------------------
#ifdef USE_TDM
__device__ __forceinline__ void tdm_issue_a_tile(const _Float16* gsrc,
                                                 unsigned lds_byte_addr) {
  unsigned long long ga = (unsigned long long)(uintptr_t)gsrc;
  u32x4 g0;
  g0[0] = 1u;                                   // count=1, user descriptor
  g0[1] = lds_byte_addr;                        // lds_addr
  g0[2] = (unsigned)(ga & 0xFFFFFFFFu);         // global_addr[31:0]
  g0[3] = (unsigned)((ga >> 32) & 0x01FFFFFFu)  // global_addr[56:32]
          | 0x80000000u;                        // type=2 ("image") bits 127:126
  i32x8 g1;
  g1[0] = 0x07510000;        // data_size=2B | pad_enable | interval=64DW | amount=4DW
  g1[1] = (int)(D_MODEL << 16);                 // tensor_dim0[15:0] @ bits 63:48
  g1[2] = (int)((unsigned)MTOK << 16);          // dim0 hi=0 | tensor_dim1[15:0]
  g1[3] = (int)(CH << 16);                      // dim1 hi=0 | tile_dim0=CH
  g1[4] = 16;                                   // tile_dim1=16 | tile_dim2=0
  g1[5] = D_MODEL;                              // tensor_dim0_stride lo
  g1[6] = 0;                                    // stride0 hi | stride1 lo
  g1[7] = 0;                                    // stride1 hi
  i32x4 g2 = {0, 0, 0, 0};
  i32x4 g3 = {0, 0, 0, 0};
#if __has_include(<hip/amd_detail/amd_gfx1250_TDM.h>)
  i32x8 g4 = {0, 0, 0, 0, 0, 0, 0, 0};          // 6-arg therock form
  __builtin_amdgcn_tensor_load_to_lds(g0, g1, g2, g3, g4, 0);
#else
  __builtin_amdgcn_tensor_load_to_lds(g0, g1, g2, g3, 0);  // 5-arg ROCm 7.2 form
#endif
}
#endif

// ---------------------------------------------------------------------------
// Shared GEMM core: C(16x64) += A(16x1024) * Wpack.  All 8 waves of the block
// share one 16-row A tile staged through LDS (TDM double-buffered); the B
// (weight) stream is register double-buffered across K-steps.
// ---------------------------------------------------------------------------
__device__ __forceinline__ void gemm_core(const _Float16* __restrict__ A,
                                          const _Float16* __restrict__ wp,
                                          int m0, int n0, int wave,
                                          _Float16* lds /* [2][16*LDSTR] */,
                                          v8f c[4]) {
#ifdef USE_TDM
  if (wave == 0)
    tdm_issue_a_tile(A + (size_t)m0 * D_MODEL, (unsigned)(uintptr_t)lds);
  for (int ch = 0; ch < NCH; ++ch) {
    _Float16* cur = lds + (ch & 1) * (16 * LDSTR);
    if (wave == 0) {
      if (ch + 1 < NCH) {
        tdm_issue_a_tile(A + (size_t)m0 * D_MODEL + (ch + 1) * CH,
                         (unsigned)(uintptr_t)(lds + ((ch + 1) & 1) * (16 * LDSTR)));
        __builtin_amdgcn_s_wait_tensorcnt(1); // chunk ch done; ch+1 in flight
      } else {
        __builtin_amdgcn_s_wait_tensorcnt(0); // last chunk fully resident
      }
    }
    __syncthreads();  // chunk ch resident in `cur`
    v16h breg[4];
#pragma unroll
    for (int t = 0; t < 4; ++t)
      breg[t] = load_b_frag_packed(wp, ch * 4, n0 + t * 16);
#pragma unroll
    for (int s = 0; s < 4; ++s) {
      v16h bnext[4];
      if (s < 3)
#pragma unroll
        for (int t = 0; t < 4; ++t)
          bnext[t] = load_b_frag_packed(wp, ch * 4 + s + 1, n0 + t * 16);
      v16h a = load_a_frag_lds(cur, s);
#pragma unroll
      for (int t = 0; t < 4; ++t) c[t] = wmma_f16(a, breg[t], c[t]);
      if (s < 3)
#pragma unroll
        for (int t = 0; t < 4; ++t) breg[t] = bnext[t];
    }
    __syncthreads();  // all waves done with `cur` before TDM overwrites it
  }
#else
  // Fallback: cooperative synchronous staging (single buffer).
  for (int ch = 0; ch < NCH; ++ch) {
    __syncthreads();
    {
      int t = threadIdx.x;          // 256 threads x 8 halves = 16x128 tile
      int row = t >> 4, col = (t & 15) * 8;
      *(v8h*)(lds + row * LDSTR + col) =
          *(const v8h*)(A + (size_t)(m0 + row) * D_MODEL + ch * CH + col);
    }
    __syncthreads();
    v16h breg[4];
#pragma unroll
    for (int t = 0; t < 4; ++t)
      breg[t] = load_b_frag_packed(wp, ch * 4, n0 + t * 16);
#pragma unroll
    for (int s = 0; s < 4; ++s) {
      v16h bnext[4];
      if (s < 3)
#pragma unroll
        for (int t = 0; t < 4; ++t)
          bnext[t] = load_b_frag_packed(wp, ch * 4 + s + 1, n0 + t * 16);
      v16h a = load_a_frag_lds(lds, s);
#pragma unroll
      for (int t = 0; t < 4; ++t) c[t] = wmma_f16(a, breg[t], c[t]);
      if (s < 3)
#pragma unroll
        for (int t = 0; t < 4; ++t) breg[t] = bnext[t];
    }
  }
#endif
}

// ---------------------------------------------------------------------------
// Converters
// ---------------------------------------------------------------------------
__global__ void cvt_x_f16(const float* __restrict__ x, _Float16* __restrict__ xh, int n) {
  int i = blockIdx.x * blockDim.x + threadIdx.x;
  if (i < n) xh[i] = (_Float16)x[i];
}

// Wpack[kt*32768 + n*32 + j] = W[n*1024 + kt*32 + j]
__global__ void pack_w(const float* __restrict__ w, _Float16* __restrict__ wp) {
  int i = blockIdx.x * blockDim.x + threadIdx.x; // 0 .. 1M-1
  int j = i & 31;
  int rest = i >> 5;
  int n = rest & (D_MODEL - 1);
  int kt = rest >> 10;
  wp[(size_t)kt * (32 * D_MODEL) + (size_t)n * 32 + j] =
      (_Float16)w[(size_t)n * D_MODEL + kt * 32 + j];
}

// ---------------------------------------------------------------------------
// Fused QKV projection. Block: 8 waves sharing token rows m0..m0+15.
//   m0 = (blockIdx.x>>1)*16  (scalar -> scalar TDM descriptor)
//   ng = 8*(blockIdx.x&1) + wave
// blockIdx.y selects Q/K/V. Q pre-scaled by 1/sqrt(dk). V stored transposed.
// ---------------------------------------------------------------------------
__global__ void __launch_bounds__(256)
qkv_gemm(const _Float16* __restrict__ xh,
         const _Float16* __restrict__ wpq, const _Float16* __restrict__ wpk,
         const _Float16* __restrict__ wpv,
         const float* __restrict__ bq, const float* __restrict__ bk,
         const float* __restrict__ bv,
         _Float16* __restrict__ Qh, _Float16* __restrict__ Kh,
         _Float16* __restrict__ Vt) {
  __shared__ __align__(16) _Float16 atile[2][16 * LDSTR];
  int wave = threadIdx.x >> 5;
  int lane = threadIdx.x & 31;
  int hl = lane >> 4, nn = lane & 15;
  int m0 = (blockIdx.x >> 1) * 16;
  int ng = 8 * (blockIdx.x & 1) + wave;
  int n0 = ng * 64;
  int z = blockIdx.y; // 0=Q 1=K 2=V

  const _Float16* wp = (z == 0) ? wpq : (z == 1) ? wpk : wpv;
  const float* bias  = (z == 0) ? bq  : (z == 1) ? bk  : bv;

  v8f c[4] = {};
  gemm_core(xh, wp, m0, n0, wave, &atile[0][0], c);

  float scale = (z == 0) ? 0.125f : 1.0f; // 1/sqrt(64) folded into Q
  int h = n0 >> 6;                        // whole 64-col group is one head
#pragma unroll
  for (int t = 0; t < 4; ++t) {
    int n = n0 + t * 16 + nn;
    float bn = bias[n];
    int d = n & 63;
#pragma unroll
    for (int r = 0; r < 8; ++r) {
      int tok = m0 + r + 8 * hl;
      int bb = tok >> 11;       // /SEQ
      int tt = tok & (SEQ - 1);
      float y = (c[t][r] + bn) * scale;
      if (z == 2) {
        Vt[(((size_t)bb * NUM_HEADS + h) * DK + d) * SEQ + tt] = (_Float16)y;
      } else {
        _Float16* dst = (z == 0) ? Qh : Kh;
        dst[(((size_t)bb * NUM_HEADS + h) * SEQ + tt) * DK + d] = (_Float16)y;
      }
    }
  }
}

// ---------------------------------------------------------------------------
// Flash attention: one wave per 16-query tile, keys streamed in 32s with
// software pipelining (V frags issued before score WMMAs; next K frags issued
// before the softmax VALU stretch).
// ---------------------------------------------------------------------------
__global__ void __launch_bounds__(256)
attn(const _Float16* __restrict__ Qh, const _Float16* __restrict__ Kh,
     const _Float16* __restrict__ Vt, _Float16* __restrict__ ctx) {
  __shared__ __align__(16) _Float16 plds[8][16 * 32]; // 1KB per wave
  int wave = threadIdx.x >> 5;
  int lane = threadIdx.x & 31;
  int hl = lane >> 4, nn = lane & 15;
  int qt = blockIdx.x * 8 + wave; // 0..127
  int bh = blockIdx.y;            // 0..31

  const _Float16* q = Qh + (size_t)bh * SEQ * DK;
  const _Float16* k = Kh + (size_t)bh * SEQ * DK;
  const _Float16* v = Vt + (size_t)bh * DK * SEQ;

  v16h qa0 = load_a_frag(q, qt * 16, DK, 0);
  v16h qa1 = load_a_frag(q, qt * 16, DK, 32);

  float run_max[8], run_sum[8];
  v8f o[4] = {};
#pragma unroll
  for (int r = 0; r < 8; ++r) { run_max[r] = -1e30f; run_sum[r] = 0.f; }

  _Float16* pl = plds[wave];

  v16h kf[4];
  kf[0] = load_b_frag_kt(k, 0, 0);
  kf[1] = load_b_frag_kt(k, 0, 32);
  kf[2] = load_b_frag_kt(k, 16, 0);
  kf[3] = load_b_frag_kt(k, 16, 32);

  for (int j0 = 0; j0 < SEQ; j0 += 32) {
    v16h vf[4];
#pragma unroll
    for (int t = 0; t < 4; ++t) vf[t] = load_b_frag_vt(v, t * 16, j0);

    v8f s0 = {}, s1 = {};
    s0 = wmma_f16(qa0, kf[0], s0);
    s0 = wmma_f16(qa1, kf[1], s0);
    s1 = wmma_f16(qa0, kf[2], s1);
    s1 = wmma_f16(qa1, kf[3], s1);

    if (j0 + 32 < SEQ) { // uniform; prefetch next key block under softmax math
      kf[0] = load_b_frag_kt(k, j0 + 32, 0);
      kf[1] = load_b_frag_kt(k, j0 + 32, 32);
      kf[2] = load_b_frag_kt(k, j0 + 48, 0);
      kf[3] = load_b_frag_kt(k, j0 + 48, 32);
    }

    // per-row max across the 16 lanes of each half-group (rows r+8*hl)
    float mx[8];
#pragma unroll
    for (int r = 0; r < 8; ++r) mx[r] = fmaxf(s0[r], s1[r]);
#pragma unroll
    for (int m = 1; m < 16; m <<= 1)
#pragma unroll
      for (int r = 0; r < 8; ++r) mx[r] = fmaxf(mx[r], __shfl_xor(mx[r], m, 32));

    float corr[8];
#pragma unroll
    for (int r = 0; r < 8; ++r) {
      float nm = fmaxf(run_max[r], mx[r]);
      corr[r] = __expf(run_max[r] - nm);
      run_max[r] = nm;
    }

    float ts[8];
#pragma unroll
    for (int r = 0; r < 8; ++r) {
      float p0 = __expf(s0[r] - run_max[r]);
      float p1 = __expf(s1[r] - run_max[r]);
      pl[(r + 8 * hl) * 32 + nn]      = (_Float16)p0; // cols 0..15  <- S0
      pl[(r + 8 * hl) * 32 + nn + 16] = (_Float16)p1; // cols 16..31 <- S1
      ts[r] = p0 + p1;
    }
#pragma unroll
    for (int m = 1; m < 16; m <<= 1)
#pragma unroll
      for (int r = 0; r < 8; ++r) ts[r] += __shfl_xor(ts[r], m, 32);
#pragma unroll
    for (int r = 0; r < 8; ++r) run_sum[r] = run_sum[r] * corr[r] + ts[r];
#pragma unroll
    for (int t = 0; t < 4; ++t)
#pragma unroll
      for (int r = 0; r < 8; ++r) o[t][r] *= corr[r];

    __syncthreads(); // cross-lane LDS stores visible before A-frag reads

    const _Float16* pr = pl + (size_t)nn * 32;
    v8h lo = *(const v8h*)(pr + 16 * hl);
    v8h hi = *(const v8h*)(pr + 16 * hl + 8);
    v16h pa;
#pragma unroll
    for (int i = 0; i < 8; ++i) { pa[i] = lo[i]; pa[i + 8] = hi[i]; }

#pragma unroll
    for (int t = 0; t < 4; ++t) o[t] = wmma_f16(pa, vf[t], o[t]);

    __syncthreads(); // reads done before next iteration's stores
  }

  int b = bh >> 4, h = bh & 15;
  float inv[8];
#pragma unroll
  for (int r = 0; r < 8; ++r) inv[r] = 1.f / run_sum[r];
#pragma unroll
  for (int t = 0; t < 4; ++t) {
    int d = t * 16 + nn;
#pragma unroll
    for (int r = 0; r < 8; ++r) {
      int tok = qt * 16 + r + 8 * hl;
      ctx[((size_t)b * SEQ + tok) * D_MODEL + h * DK + d] =
          (_Float16)(o[t][r] * inv[r]);
    }
  }
}

// ---------------------------------------------------------------------------
// Output projection: out = ctx @ Wo^T + bo (f32 epilogue straight to d_out)
// ---------------------------------------------------------------------------
__global__ void __launch_bounds__(256)
out_gemm(const _Float16* __restrict__ ctx, const _Float16* __restrict__ wpo,
         const float* __restrict__ bo, float* __restrict__ out) {
  __shared__ __align__(16) _Float16 atile[2][16 * LDSTR];
  int wave = threadIdx.x >> 5;
  int lane = threadIdx.x & 31;
  int hl = lane >> 4, nn = lane & 15;
  int m0 = (blockIdx.x >> 1) * 16;
  int n0 = (8 * (blockIdx.x & 1) + wave) * 64;

  v8f c[4] = {};
  gemm_core(ctx, wpo, m0, n0, wave, &atile[0][0], c);

#pragma unroll
  for (int t = 0; t < 4; ++t) {
    int n = n0 + t * 16 + nn;
    float bn = bo[n];
#pragma unroll
    for (int r = 0; r < 8; ++r)
      out[(size_t)(m0 + r + 8 * hl) * D_MODEL + n] = c[t][r] + bn;
  }
}

// ---------------------------------------------------------------------------
// Host entry. Workspace layout (bytes):
//   xh 0..8M | WpQ 8M | WpK 10M | WpV 12M | WpO 14M
//   Qh 16M | Kh 24M | Vt 32M | ctx 40M   (total 48MB)
// ---------------------------------------------------------------------------
extern "C" void kernel_launch(void* const* d_in, const int* in_sizes, int n_in,
                              void* d_out, int out_size, void* d_ws, size_t ws_size,
                              hipStream_t stream) {
  (void)in_sizes; (void)n_in; (void)out_size; (void)ws_size;
  const float* x  = (const float*)d_in[0];
  const float* Wq = (const float*)d_in[1];
  const float* bq = (const float*)d_in[2];
  const float* Wk = (const float*)d_in[3];
  const float* bk = (const float*)d_in[4];
  const float* Wv = (const float*)d_in[5];
  const float* bv = (const float*)d_in[6];
  const float* Wo = (const float*)d_in[7];
  const float* bo = (const float*)d_in[8];
  float* out = (float*)d_out;

  char* ws = (char*)d_ws;
  const size_t MB = 1024 * 1024;
  _Float16* xh  = (_Float16*)(ws + 0);
  _Float16* wpq = (_Float16*)(ws + 8 * MB);
  _Float16* wpk = (_Float16*)(ws + 10 * MB);
  _Float16* wpv = (_Float16*)(ws + 12 * MB);
  _Float16* wpo = (_Float16*)(ws + 14 * MB);
  _Float16* Qh  = (_Float16*)(ws + 16 * MB);
  _Float16* Kh  = (_Float16*)(ws + 24 * MB);
  _Float16* Vt  = (_Float16*)(ws + 32 * MB);
  _Float16* ctx = (_Float16*)(ws + 40 * MB);

  const int NX = MTOK * D_MODEL;    // 4M
  const int NW = D_MODEL * D_MODEL; // 1M

  cvt_x_f16<<<(NX + 255) / 256, 256, 0, stream>>>(x, xh, NX);
  pack_w<<<(NW + 255) / 256, 256, 0, stream>>>(Wq, wpq);
  pack_w<<<(NW + 255) / 256, 256, 0, stream>>>(Wk, wpk);
  pack_w<<<(NW + 255) / 256, 256, 0, stream>>>(Wv, wpv);
  pack_w<<<(NW + 255) / 256, 256, 0, stream>>>(Wo, wpo);

  qkv_gemm<<<dim3(512, 3), 256, 0, stream>>>(xh, wpq, wpk, wpv, bq, bk, bv,
                                             Qh, Kh, Vt);
  attn<<<dim3(16, 32), 256, 0, stream>>>(Qh, Kh, Vt, ctx);
  out_gemm<<<512, 256, 0, stream>>>(ctx, wpo, bo, out);
}
